// GPTOSSTransformerBlock_73521250173673
// MI455X (gfx1250) — compile-verified, matrix operand order checked by reading, not run on hardware
//
#include <hip/hip_runtime.h>

typedef __attribute__((ext_vector_type(16))) __bf16 v16bf;
typedef __attribute__((ext_vector_type(8)))  __bf16 v8bf;
typedef __attribute__((ext_vector_type(8)))  float  v8f;

#define T_TOK 4096   // B*S
#define H_DIM 2048
#define I_DIM 4096
#define N_H   16
#define N_KV  4
#define H_D   128
#define S_LEN 2048
#define N_E   8
#define E_CAP 4096

static __device__ __forceinline__ v8f wmma_bf16(v16bf a, v16bf b, v8f c) {
  // (neg_a, A, neg_b, B, c_mod, C, reuse_a, reuse_b)
  return __builtin_amdgcn_wmma_f32_16x16x32_bf16(false, a, false, b, (short)0, c,
                                                 false, false);
}

// Fragment from a bf16 LDS tile laid out so the K-runs are contiguous:
// lane&15 selects the "row" (A: M row; B: N column of a transposed tile);
// elements 0..7 = cols [g*8, g*8+8), 8..15 = cols [g*8+16, g*8+24).
// Both runs are 16B -> two ds_load_b128, zero conversions.
static __device__ __forceinline__ v16bf frag_bf16(const __bf16* src, int ld) {
  int lane = threadIdx.x & 31;
  int g = lane >> 4;
  int m = lane & 15;
  const __bf16* row = src + m * ld + g * 8;
  v8bf lo = *(const v8bf*)row;
  v8bf hi = *(const v8bf*)(row + 16);
  v16bf f;
#pragma unroll
  for (int e = 0; e < 8; ++e) { f[e] = lo[e]; f[e + 8] = hi[e]; }
  return f;
}

static __device__ __forceinline__ float redmax16(float v) {
  v = fmaxf(v, __shfl_xor(v, 1, 32));
  v = fmaxf(v, __shfl_xor(v, 2, 32));
  v = fmaxf(v, __shfl_xor(v, 4, 32));
  v = fmaxf(v, __shfl_xor(v, 8, 32));
  return v;
}
static __device__ __forceinline__ float redsum16(float v) {
  v += __shfl_xor(v, 1, 32);
  v += __shfl_xor(v, 2, 32);
  v += __shfl_xor(v, 4, 32);
  v += __shfl_xor(v, 8, 32);
  return v;
}

// Async global->LDS copy (16B per lane), tracked by ASYNCcnt.
static __device__ __forceinline__ void async_ld_b128(unsigned lds_addr,
                                                     unsigned goff,
                                                     unsigned long long base) {
  asm volatile("global_load_async_to_lds_b128 %0, %1, %2"
               :: "v"(lds_addr), "v"(goff), "s"(base) : "memory");
}
static __device__ __forceinline__ void wait_async0() {
  asm volatile("s_wait_asynccnt 0" ::: "memory");
}

// ---------------------------------------------------------------------------
// WMMA GEMM: C[MxN] = A[MxK] @ B[KxN], row-major f32 in HBM, bf16 math.
// Block 256 = 8 waves -> 32x128 C tile; each wave: one A-frag, two B-frags,
// two WMMAs (16x32 C slice). LDS tiles stored in bf16; B transposed so all
// fragment reads are contiguous b128 pairs (convert-once at staging).
// MODE 0: C = acc ; MODE 1: C = acc + R ; MODE 2: C[rowmap[row]] += roww*acc
// GATHER: A row index from rowmap, bounded by cntp[0].
// ---------------------------------------------------------------------------
template <int MODE, bool GATHER>
__global__ void __launch_bounds__(256)
gemm_wmma(const float* __restrict__ A, const float* __restrict__ B,
          float* __restrict__ C, int M, int N, int K,
          const float* __restrict__ R, const int* __restrict__ rowmap,
          const float* __restrict__ roww, const int* __restrict__ cntp)
{
  __shared__ alignas(16) __bf16 As[32 * 32];    // [m][k] bf16
  __shared__ alignas(16) __bf16 Bt[128 * 32];   // [n][k] bf16 (transposed)
  const int cnt = cntp ? cntp[0] : M;
  const int mblk = blockIdx.y * 32;
  if (mblk >= cnt) return;          // block-uniform early exit (before barriers)
  const int nblk = blockIdx.x * 128;
  const int tid  = threadIdx.x;
  const int wave = tid >> 5;
  const int lane = tid & 31;
  const int wx = wave & 3;          // n direction: 32 cols per wave
  const int wy = wave >> 2;         // m direction: 16 rows per wave
  const int g   = lane >> 4;
  const int n16 = lane & 15;

  v8f acc0 = {}, acc1 = {};
  for (int k0 = 0; k0 < K; k0 += 32) {
    __syncthreads();
    {   // A tile 32x32: 4 consecutive f32 per thread -> cvt -> bf16 store
      int base = tid * 4;
      int r = base >> 5, c = base & 31;
      int row = mblk + r;
      float4 va = make_float4(0.f, 0.f, 0.f, 0.f);
      if (row < cnt) {
        int srow = GATHER ? rowmap[row] : row;
        va = *(const float4*)(A + (size_t)srow * K + k0 + c);
      }
      As[base + 0] = (__bf16)va.x;
      As[base + 1] = (__bf16)va.y;
      As[base + 2] = (__bf16)va.z;
      As[base + 3] = (__bf16)va.w;
    }
    {   // B tile 32x128: 16 consecutive f32 per thread, store transposed
      int base = tid * 16;
      int r = base >> 7, c0 = base & 127;
      const float* bp = B + (size_t)(k0 + r) * N + nblk + c0;
#pragma unroll
      for (int j = 0; j < 16; ++j)
        Bt[(c0 + j) * 32 + r] = (__bf16)bp[j];
    }
    if (k0 + 32 < K)                            // global_prefetch_b8 next tile
      __builtin_prefetch(B + (size_t)(k0 + 32) * N + nblk + (tid & 127), 0, 3);
    __syncthreads();
    v16bf fa  = frag_bf16(As + wy * 16 * 32, 32);
    v16bf fb0 = frag_bf16(Bt + (wx * 32) * 32, 32);
    v16bf fb1 = frag_bf16(Bt + (wx * 32 + 16) * 32, 32);
    acc0 = wmma_bf16(fa, fb0, acc0);
    acc1 = wmma_bf16(fa, fb1, acc1);
  }

  const int col0 = nblk + wx * 32 + n16;
#pragma unroll
  for (int r = 0; r < 8; ++r) {
    int row = mblk + wy * 16 + r + g * 8;       // C layout: VGPR r -> row r+8g
    if (row >= cnt) continue;
    float v0 = acc0[r], v1 = acc1[r];
    if (MODE == 0) {
      C[(size_t)row * N + col0]      = v0;
      C[(size_t)row * N + col0 + 16] = v1;
    } else if (MODE == 1) {
      C[(size_t)row * N + col0]      = v0 + R[(size_t)row * N + col0];
      C[(size_t)row * N + col0 + 16] = v1 + R[(size_t)row * N + col0 + 16];
    } else {
      int t = rowmap[row];
      C[(size_t)t * N + col0]      += roww[row] * v0;
      C[(size_t)t * N + col0 + 16] += roww[row] * v1;
    }
  }
}

// ---------------------------------------------------------------------------
// Flash attention: grid (S/64, NH, B), 128 threads = 4 waves x 16 query rows.
// K/V chunks (32 keys x 128d) pulled into LDS with GLOBAL_LOAD_ASYNC_TO_LDS
// (ASYNCcnt), then converted once to bf16 (K row-major, V transposed) so all
// WMMA fragments are contiguous b128 LDS reads. Online softmax; P re-laid out
// through per-wave bf16 LDS scratch for the P@V WMMA.
// ---------------------------------------------------------------------------
__global__ void __launch_bounds__(128)
attn_kernel(const float* __restrict__ Q, const float* __restrict__ Kb,
            const float* __restrict__ Vb, float* __restrict__ O)
{
  __shared__ alignas(16) float  Kst[32 * H_D];   // async landing pads (f32)
  __shared__ alignas(16) float  Vst[32 * H_D];
  __shared__ alignas(16) __bf16 Ksb[32 * H_D];   // [key][d] bf16
  __shared__ alignas(16) __bf16 Vsb[H_D * 32];   // [d][key] bf16 (transposed)
  __shared__ alignas(16) __bf16 Ps[4][16 * 32];  // per-wave P scratch
  const int qblk = blockIdx.x * 64;
  const int h    = blockIdx.y;
  const int b    = blockIdx.z;
  const int kvh  = h / (N_H / N_KV);
  const int tid  = threadIdx.x;
  const int wave = tid >> 5;
  const int lane = tid & 31;
  const int g    = lane >> 4;
  const int idx  = lane & 15;
  const int q0   = qblk + wave * 16;

  const float* Qb = Q  + (size_t)b * S_LEN * (N_H * H_D)  + (size_t)h * H_D;
  const float* Kp = Kb + (size_t)b * S_LEN * (N_KV * H_D) + (size_t)kvh * H_D;
  const float* Vp = Vb + (size_t)b * S_LEN * (N_KV * H_D) + (size_t)kvh * H_D;
  const unsigned ldsK = (unsigned)(uintptr_t)&Kst[0];
  const unsigned ldsV = (unsigned)(uintptr_t)&Vst[0];
  const unsigned long long baseK = (unsigned long long)(uintptr_t)Kp;
  const unsigned long long baseV = (unsigned long long)(uintptr_t)Vp;

  const float scale = 0.08838834764831845f;     // 1/sqrt(128), folded into Q
  v16bf qf[4];                                  // Q 16x128 as 4 A-frags
  {
    const float* qrow = Qb + (size_t)(q0 + idx) * (N_H * H_D);
#pragma unroll
    for (int c = 0; c < 4; ++c) {
      const float* p = qrow + c * 32 + g * 8;
      v16bf f;
#pragma unroll
      for (int e = 0; e < 8; ++e)  f[e] = (__bf16)(p[e] * scale);
#pragma unroll
      for (int e = 8; e < 16; ++e) f[e] = (__bf16)(p[e + 8] * scale);
      qf[c] = f;
    }
  }

  float mrow[8], lrow[8];
  v8f o[8] = {};
#pragma unroll
  for (int r = 0; r < 8; ++r) { mrow[r] = -1e30f; lrow[r] = 0.0f; }

  const int kend = qblk + 64;
  for (int kc = 0; kc < kend; kc += 32) {
    __syncthreads();
    // Async global->LDS: 32x128 f32 K and V chunks, 16B per lane per issue.
#pragma unroll
    for (int i = 0; i < 8; ++i) {
      int u = tid + i * 128;                   // 16B-unit index, 0..1023
      int r  = u >> 5;                         // key row (32 x 16B per row)
      int cb = (u & 31) << 4;                  // byte offset within row
      unsigned go = (unsigned)((kc + r) * (N_KV * H_D * 4) + cb);
      unsigned lo = (unsigned)(u << 4);
      async_ld_b128(ldsK + lo, go, baseK);
      async_ld_b128(ldsV + lo, go, baseV);
    }
    wait_async0();
    __syncthreads();
    // Convert once: K -> bf16 row-major, V -> bf16 transposed [d][key].
#pragma unroll
    for (int i = 0; i < 16; ++i) {
      int e2 = (tid + i * 128) * 2;
      Ksb[e2]     = (__bf16)Kst[e2];
      Ksb[e2 + 1] = (__bf16)Kst[e2 + 1];
    }
#pragma unroll
    for (int k2 = 0; k2 < 32; ++k2)            // this thread owns column d=tid
      Vsb[tid * 32 + k2] = (__bf16)Vst[k2 * H_D + tid];
    __syncthreads();

    const bool active = (kc <= q0 + 15);       // wave-uniform causal skip
    if (active) {
      v8f s0 = {}, s1 = {};                    // two 16-key score tiles
#pragma unroll
      for (int c = 0; c < 4; ++c) {
        v16bf b0 = frag_bf16(Ksb + c * 32, H_D);            // keys 0..15
        v16bf b1 = frag_bf16(Ksb + 16 * H_D + c * 32, H_D); // keys 16..31
        s0 = wmma_bf16(qf[c], b0, s0);
        s1 = wmma_bf16(qf[c], b1, s1);
      }
      __bf16* P = &Ps[wave][0];
#pragma unroll
      for (int r = 0; r < 8; ++r) {            // online softmax per row r+8g
        const int qrow = q0 + r + g * 8;
        float v0 = (kc + idx      <= qrow) ? s0[r] : -1e30f;
        float v1 = (kc + 16 + idx <= qrow) ? s1[r] : -1e30f;
        float mc    = redmax16(fmaxf(v0, v1));
        float mnew  = fmaxf(mrow[r], mc);
        float alpha = __expf(mrow[r] - mnew);
        mrow[r] = mnew;
        float p0 = __expf(v0 - mnew);
        float p1 = __expf(v1 - mnew);
        lrow[r] = lrow[r] * alpha + redsum16(p0 + p1);
#pragma unroll
        for (int dt = 0; dt < 8; ++dt) o[dt][r] *= alpha;
        P[(r + g * 8) * 32 + idx]      = (__bf16)p0;
        P[(r + g * 8) * 32 + 16 + idx] = (__bf16)p1;
      }
    }
    __syncthreads();
    if (active) {
      v16bf pf = frag_bf16(&Ps[wave][0], 32);  // P as A-frag (16q x 32keys)
#pragma unroll
      for (int dt = 0; dt < 8; ++dt) {
        v16bf vf = frag_bf16(Vsb + dt * 16 * 32, 32);       // d cols dt*16..
        o[dt] = wmma_bf16(pf, vf, o[dt]);
      }
    }
  }

#pragma unroll
  for (int r = 0; r < 8; ++r) {
    const int qrow = q0 + r + g * 8;
    const float inv = 1.0f / lrow[r];
    float* orow = O + (size_t)(b * S_LEN + qrow) * (N_H * H_D) + (size_t)h * H_D;
#pragma unroll
    for (int dt = 0; dt < 8; ++dt)
      orow[dt * 16 + idx] = o[dt][r] * inv;
  }
}

// ---------------------------------------------------------------------------
__global__ void __launch_bounds__(256)
rmsnorm_kernel(const float* __restrict__ x, const float* __restrict__ w,
               float* __restrict__ out)
{
  __shared__ float red[256];
  const size_t base = (size_t)blockIdx.x * H_DIM;
  const int tid = threadIdx.x;
  float ss = 0.0f;
#pragma unroll
  for (int i = 0; i < 8; ++i) {
    float v = x[base + tid + i * 256];
    ss += v * v;
  }
  red[tid] = ss;
  __syncthreads();
  for (int s = 128; s > 0; s >>= 1) {
    if (tid < s) red[tid] += red[tid + s];
    __syncthreads();
  }
  const float inv = rsqrtf(red[0] * (1.0f / (float)H_DIM) + 1e-5f);
#pragma unroll
  for (int i = 0; i < 8; ++i) {
    int hh = tid + i * 256;
    out[base + hh] = x[base + hh] * inv * w[hh];
  }
}

__global__ void rope_kernel(float* __restrict__ buf, int heads, int total)
{
  int gid = blockIdx.x * 256 + threadIdx.x;
  if (gid >= total) return;
  int d    = gid & 63;
  int rest = gid >> 6;
  int head = rest % heads;
  int t    = rest / heads;
  int s    = t & (S_LEN - 1);
  float ang = (float)s * __powf(10000.0f, -(float)d * (1.0f / 64.0f));
  float cs = __cosf(ang), sn = __sinf(ang);
  float* row = buf + ((size_t)t * heads + head) * H_D;
  float x1 = row[d], x2 = row[d + 64];
  row[d]      = x1 * cs - x2 * sn;
  row[d + 64] = x2 * cs + x1 * sn;
}

__global__ void __launch_bounds__(256)
router_kernel(const float* __restrict__ X, const float* __restrict__ W,
              int* __restrict__ topi, float* __restrict__ topw)
{
  __shared__ float part[256 * 8];
  __shared__ float logits[8];
  const int t = blockIdx.x;
  const int tid = threadIdx.x;
  float acc[8];
#pragma unroll
  for (int e = 0; e < 8; ++e) acc[e] = 0.0f;
  const float* xrow = X + (size_t)t * H_DIM;
#pragma unroll
  for (int i = 0; i < 8; ++i) {
    int hh = tid * 8 + i;
    float xv = xrow[hh];
#pragma unroll
    for (int e = 0; e < 8; ++e) acc[e] += xv * W[hh * 8 + e];
  }
#pragma unroll
  for (int e = 0; e < 8; ++e) part[tid * 8 + e] = acc[e];
  __syncthreads();
  if (tid < 8) {
    float s = 0.0f;
    for (int i = 0; i < 256; ++i) s += part[i * 8 + tid];
    logits[tid] = s;
  }
  __syncthreads();
  if (tid == 0) {
    float mx = logits[0];
    for (int e = 1; e < 8; ++e) mx = fmaxf(mx, logits[e]);
    float p[8], sum = 0.0f;
    for (int e = 0; e < 8; ++e) { p[e] = __expf(logits[e] - mx); sum += p[e]; }
    for (int e = 0; e < 8; ++e) p[e] /= sum;
    int e1 = 0;
    for (int e = 1; e < 8; ++e) if (p[e] > p[e1]) e1 = e;
    int e2 = (e1 == 0) ? 1 : 0;
    for (int e = 0; e < 8; ++e) if (e != e1 && p[e] > p[e2]) e2 = e;
    float w1 = p[e1], w2 = p[e2], wsum = w1 + w2;
    topi[t * 2 + 0] = e1;        topi[t * 2 + 1] = e2;
    topw[t * 2 + 0] = w1 / wsum; topw[t * 2 + 1] = w2 / wsum;
  }
}

__global__ void zero_f32_kernel(float* __restrict__ p, int n) {
  int gid = blockIdx.x * 256 + threadIdx.x;
  if (gid < n) p[gid] = 0.0f;
}
__global__ void zero_counts_kernel(int* __restrict__ c) {
  if (threadIdx.x < N_E) c[threadIdx.x] = 0;
}
__global__ void assign_kernel(const int* __restrict__ topi,
                              const float* __restrict__ topw,
                              int* __restrict__ counts, int* __restrict__ rowmap,
                              float* __restrict__ roww)
{
  int t = blockIdx.x * 256 + threadIdx.x;
  if (t >= T_TOK) return;
#pragma unroll
  for (int j = 0; j < 2; ++j) {
    int e = topi[t * 2 + j];
    int slot = atomicAdd(&counts[e], 1);
    rowmap[e * E_CAP + slot] = t;
    roww[e * E_CAP + slot]   = topw[t * 2 + j];
  }
}
__global__ void silu_mul_kernel(float* __restrict__ gbuf,
                                const float* __restrict__ ubuf,
                                const int* __restrict__ cntp)
{
  int gid = blockIdx.x * 256 + threadIdx.x;
  int row = gid >> 12;                         // I_DIM = 4096 cols per row
  if (row >= cntp[0]) return;
  float gv = gbuf[gid];
  float sv = gv / (1.0f + __expf(-gv));
  gbuf[gid] = sv * ubuf[gid];
}
__global__ void add_kernel(const float* __restrict__ a, const float* __restrict__ b,
                           float* __restrict__ out, int n)
{
  int gid = blockIdx.x * 256 + threadIdx.x;
  if (gid < n) out[gid] = a[gid] + b[gid];
}

// ---------------------------------------------------------------------------
extern "C" void kernel_launch(void* const* d_in, const int* in_sizes, int n_in,
                              void* d_out, int out_size, void* d_ws, size_t ws_size,
                              hipStream_t stream)
{
  (void)in_sizes; (void)n_in; (void)out_size; (void)ws_size;
  const float* hidden = (const float*)d_in[0];
  const float* ln1    = (const float*)d_in[1];
  const float* ln2    = (const float*)d_in[2];
  const float* wq     = (const float*)d_in[3];
  const float* wk     = (const float*)d_in[4];
  const float* wv     = (const float*)d_in[5];
  const float* wo     = (const float*)d_in[6];
  const float* rw     = (const float*)d_in[7];
  const float* gw     = (const float*)d_in[8];
  const float* uw     = (const float*)d_in[9];
  const float* dwn    = (const float*)d_in[10];
  float* out = (float*)d_out;

  float* ws = (float*)d_ws;
  float* xn    = ws;  ws += (size_t)T_TOK * H_DIM;
  float* qb    = ws;  ws += (size_t)T_TOK * H_DIM;
  float* kb    = ws;  ws += (size_t)T_TOK * (N_KV * H_D);
  float* vb    = ws;  ws += (size_t)T_TOK * (N_KV * H_D);
  float* attn  = ws;  ws += (size_t)T_TOK * H_DIM;
  float* hid2  = ws;  ws += (size_t)T_TOK * H_DIM;
  float* xn2   = ws;  ws += (size_t)T_TOK * H_DIM;
  float* moe   = ws;  ws += (size_t)T_TOK * H_DIM;
  float* gbuf  = ws;  ws += (size_t)T_TOK * I_DIM;
  float* ubuf  = ws;  ws += (size_t)T_TOK * I_DIM;
  float* rowwf = ws;  ws += (size_t)N_E * E_CAP;
  float* topw  = ws;  ws += (size_t)T_TOK * 2;
  int* topi    = (int*)ws;
  int* counts  = topi + T_TOK * 2;
  int* rowmap  = counts + 64;

  // LN1 + QKV projections + RoPE
  rmsnorm_kernel<<<T_TOK, 256, 0, stream>>>(hidden, ln1, xn);
  gemm_wmma<0, false><<<dim3(H_DIM / 128, T_TOK / 32), 256, 0, stream>>>(
      xn, wq, qb, T_TOK, H_DIM, H_DIM, nullptr, nullptr, nullptr, nullptr);
  gemm_wmma<0, false><<<dim3((N_KV * H_D) / 128, T_TOK / 32), 256, 0, stream>>>(
      xn, wk, kb, T_TOK, N_KV * H_D, H_DIM, nullptr, nullptr, nullptr, nullptr);
  gemm_wmma<0, false><<<dim3((N_KV * H_D) / 128, T_TOK / 32), 256, 0, stream>>>(
      xn, wv, vb, T_TOK, N_KV * H_D, H_DIM, nullptr, nullptr, nullptr, nullptr);
  {
    int totq = T_TOK * N_H * 64;
    rope_kernel<<<totq / 256, 256, 0, stream>>>(qb, N_H, totq);
    int totk = T_TOK * N_KV * 64;
    rope_kernel<<<totk / 256, 256, 0, stream>>>(kb, N_KV, totk);
  }

  // Flash attention + output projection (+residual)
  attn_kernel<<<dim3(S_LEN / 64, N_H, 2), 128, 0, stream>>>(qb, kb, vb, attn);
  gemm_wmma<1, false><<<dim3(H_DIM / 128, T_TOK / 32), 256, 0, stream>>>(
      attn, wo, hid2, T_TOK, H_DIM, H_DIM, hidden, nullptr, nullptr, nullptr);

  // LN2 + router top-2 + expert assignment
  rmsnorm_kernel<<<T_TOK, 256, 0, stream>>>(hid2, ln2, xn2);
  router_kernel<<<T_TOK, 256, 0, stream>>>(xn2, rw, topi, topw);
  zero_f32_kernel<<<(T_TOK * H_DIM) / 256, 256, 0, stream>>>(moe, T_TOK * H_DIM);
  zero_counts_kernel<<<1, 32, 0, stream>>>(counts);
  assign_kernel<<<T_TOK / 256, 256, 0, stream>>>(topi, topw, counts, rowmap, rowwf);

  // Per-expert gathered GEMMs (stream-ordered, so gbuf/ubuf are reused safely)
  for (int e = 0; e < N_E; ++e) {
    const float* ge = gw  + (size_t)e * H_DIM * I_DIM;
    const float* ue = uw  + (size_t)e * H_DIM * I_DIM;
    const float* de = dwn + (size_t)e * I_DIM * H_DIM;
    const int*   cm = rowmap + e * E_CAP;
    const float* cw = rowwf + e * E_CAP;
    const int*   cp = counts + e;
    gemm_wmma<0, true><<<dim3(I_DIM / 128, E_CAP / 32), 256, 0, stream>>>(
        xn2, ge, gbuf, E_CAP, I_DIM, H_DIM, nullptr, cm, nullptr, cp);
    gemm_wmma<0, true><<<dim3(I_DIM / 128, E_CAP / 32), 256, 0, stream>>>(
        xn2, ue, ubuf, E_CAP, I_DIM, H_DIM, nullptr, cm, nullptr, cp);
    silu_mul_kernel<<<(E_CAP * I_DIM) / 256, 256, 0, stream>>>(gbuf, ubuf, cp);
    gemm_wmma<2, false><<<dim3(H_DIM / 128, E_CAP / 32), 256, 0, stream>>>(
        gbuf, de, moe, E_CAP, H_DIM, I_DIM, nullptr, cm, cw, cp);
  }

  // Final residual add
  add_kernel<<<(T_TOK * H_DIM) / 256, 256, 0, stream>>>(hid2, moe, out,
                                                        T_TOK * H_DIM);
}
